// TransformerBlock_19035295056092
// MI455X (gfx1250) — matrix operand order repaired and checked by reading, not compiled, not run on hardware
//
#include <hip/hip_runtime.h>
#include <hip/hip_bf16.h>

// Problem constants (from reference): B=2, S=2048, D=1024, H=16, DH=64, DF=4096
#define BB   2
#define SS   2048
#define DDIM 1024
#define HH   16
#define DHH  64
#define DFF  4096

typedef __bf16 bf16_t;
typedef bf16_t         v16bf __attribute__((ext_vector_type(16)));
typedef float          v8f   __attribute__((ext_vector_type(8)));
typedef unsigned int   v4u   __attribute__((ext_vector_type(4)));
typedef unsigned short v4us  __attribute__((ext_vector_type(4)));

union Frag {
    v16bf v;
    unsigned short h[16];
    unsigned int   u[8];
    v4u            q[2];
};

__device__ __forceinline__ unsigned short f2bfbits(float f) {
    union { float f; unsigned u; } x; x.f = f;
    unsigned r = x.u + 0x7FFFu + ((x.u >> 16) & 1u);   // round-to-nearest-even
    return (unsigned short)(r >> 16);
}

// ---------------------------------------------------------------------------
// Weight conversion: fp32 -> bf16 bits, TRANSPOSED to [N,K] so WMMA B-fragments
// are two contiguous 16-byte k reads per lane.
// ---------------------------------------------------------------------------
__global__ void cvt_headw_t(const float* __restrict__ W, unsigned short* __restrict__ O) {
    long i = (long)blockIdx.x * blockDim.x + threadIdx.x;
    if (i < (long)HH * DDIM * DHH) {
        int  k = (int)(i % DHH);
        int  d = (int)((i / DHH) % DDIM);
        int  h = (int)(i / ((long)DHH * DDIM));
        O[(long)(h * DHH + k) * DDIM + d] = f2bfbits(W[i]);
    }
}

__global__ void cvt_w_t(const float* __restrict__ W, unsigned short* __restrict__ O,
                        int K, int N) {
    long i = (long)blockIdx.x * blockDim.x + threadIdx.x;
    if (i < (long)K * N) {
        int n = (int)(i % N);
        int k = (int)(i / N);
        O[(long)n * K + k] = f2bfbits(W[i]);
    }
}

// ---------------------------------------------------------------------------
// LayerNorm: one block per row of D=1024, vectorized (1 float4 per thread).
// ---------------------------------------------------------------------------
__global__ __launch_bounds__(256) void layernorm_k(
    const float* __restrict__ X, const float* __restrict__ g,
    const float* __restrict__ off, unsigned short* __restrict__ out)
{
    __shared__ float red[256];
    __shared__ float red2[256];
    const int row = blockIdx.x;
    const int t   = threadIdx.x;
    const float4 x = ((const float4*)(X + (long)row * DDIM))[t];
    red[t]  = x.x + x.y + x.z + x.w;
    red2[t] = x.x * x.x + x.y * x.y + x.z * x.z + x.w * x.w;
    __syncthreads();
    for (int w = 128; w > 0; w >>= 1) {
        if (t < w) { red[t] += red[t + w]; red2[t] += red2[t + w]; }
        __syncthreads();
    }
    const float mean = red[0] * (1.f / DDIM);
    const float var  = fmaxf(red2[0] * (1.f / DDIM) - mean * mean, 0.f);
    const float inv  = 1.f / (sqrtf(var) + 1e-9f);
    const float4 gg = ((const float4*)g)[t];
    const float4 oo = ((const float4*)off)[t];
    v4us o4;
    o4[0] = f2bfbits(gg.x * ((x.x - mean) * inv) + oo.x);
    o4[1] = f2bfbits(gg.y * ((x.y - mean) * inv) + oo.y);
    o4[2] = f2bfbits(gg.z * ((x.z - mean) * inv) + oo.z);
    o4[3] = f2bfbits(gg.w * ((x.w - mean) * inv) + oo.w);
    *(v4us*)(out + (long)row * DDIM + t * 4) = o4;
}

// ---------------------------------------------------------------------------
// bf16 GEMM (TN): C[M,N] = A[M,K] * Bt[N,K]^T. One wave -> 32x64 tile,
// 4 waves/block share one 64x32 B tile staged in LDS by the CDNA5 async
// copy engine (global_load_async_to_lds_b128, ASYNCcnt), double buffered.
// ---------------------------------------------------------------------------
template<int RELU, int HAS_RES, int OUTF, int OUTB, int OUTBT>
__global__ __launch_bounds__(128) void gemm_tn(
    const unsigned short* __restrict__ A, const unsigned short* __restrict__ Bt,
    const float* __restrict__ bias, const float* __restrict__ resid,
    float* __restrict__ outF, unsigned short* __restrict__ outB,
    unsigned short* __restrict__ outBT,
    int M, int N, int K)
{
    __shared__ __align__(16) unsigned short Bsh[2][64 * 32];  // [buf][n(64) x k(32)]

    const int t     = threadIdx.x;
    const int lane  = t & 31;
    const int wave  = t >> 5;
    const int row16 = lane & 15;
    const int grp   = lane >> 4;
    const int m0 = (blockIdx.y * 4 + wave) * 32;
    const int n0 = blockIdx.x * 64;

    // async-copy assignment: thread t moves half a tile row (32B) per chunk
    const int crow  = t >> 1;        // 0..63  (n within tile)
    const int chalf = t & 1;         // 0/1    (k half of the 32-wide chunk)
    const unsigned short* gB = Bt + (long)(n0 + crow) * K + chalf * 16;

    auto issue_copy = [&](int buf, int kk) {
        unsigned lds0 = (unsigned)(unsigned long long)&Bsh[buf][crow * 32 + chalf * 16];
        unsigned long long gaddr = (unsigned long long)(gB + kk);
        asm volatile("global_load_async_to_lds_b128 %0, %1, off"
                     :: "v"(lds0), "v"(gaddr) : "memory");
    };

    v8f acc[8] = {};   // [mi*4 + jn]
    const unsigned short* ap0 = A + (long)(m0 + row16) * K + 8 * grp;
    const unsigned short* ap1 = ap0 + (long)16 * K;

    issue_copy(0, 0);
    for (int kk = 0; kk < K; kk += 32) {
        const int buf = (kk >> 5) & 1;
        asm volatile("s_wait_asynccnt 0" ::: "memory");  // our copies landed in LDS
        __syncthreads();                                 // all waves' copies visible
        if (kk + 32 < K) issue_copy(buf ^ 1, kk + 32);   // overlap next copy w/ math

        Frag a0, a1;
        a0.q[0] = *(const v4u*)(ap0 + kk);
        a0.q[1] = *(const v4u*)(ap0 + kk + 16);
        a1.q[0] = *(const v4u*)(ap1 + kk);
        a1.q[1] = *(const v4u*)(ap1 + kk + 16);
#pragma unroll
        for (int jn = 0; jn < 4; ++jn) {
            Frag bf;
            const unsigned short* bp = &Bsh[buf][(jn * 16 + row16) * 32 + 8 * grp];
            bf.q[0] = *(const v4u*)(bp);
            bf.q[1] = *(const v4u*)(bp + 16);
            acc[0 * 4 + jn] = __builtin_amdgcn_wmma_f32_16x16x32_bf16(
                false, a0.v, false, bf.v, (short)0, acc[0 * 4 + jn], false, false);
            acc[1 * 4 + jn] = __builtin_amdgcn_wmma_f32_16x16x32_bf16(
                false, a1.v, false, bf.v, (short)0, acc[1 * 4 + jn], false, false);
        }
    }

    float bcol[4];
#pragma unroll
    for (int jn = 0; jn < 4; ++jn) bcol[jn] = bias[n0 + jn * 16 + row16];

#pragma unroll
    for (int mi = 0; mi < 2; ++mi) {
#pragma unroll
        for (int jn = 0; jn < 4; ++jn) {
            const int n = n0 + jn * 16 + row16;
#pragma unroll
            for (int v = 0; v < 8; ++v) {
                const int m = m0 + mi * 16 + v + 8 * grp;
                float val = acc[mi * 4 + jn][v] + bcol[jn];
                if constexpr (RELU) val = fmaxf(val, 0.f);
                const long idx = (long)m * N + n;
                if constexpr (HAS_RES) val += resid[idx];
                if constexpr (OUTF)  outF[idx] = val;
                if constexpr (OUTB)  outB[idx] = f2bfbits(val);
                if constexpr (OUTBT) {
                    const int bb = m >> 11;        // S = 2048
                    const int s  = m & 2047;
                    outBT[((long)bb * DDIM + n) * SS + s] = f2bfbits(val);
                }
            }
        }
    }
}

// ---------------------------------------------------------------------------
// Causal flash attention. Q,K: [B,S,H*DH] bf16; Vt: [B,H*DH,S] bf16.
// grid = (S/64, H, B), block = 128 (4 waves, one 16-query tile each).
// tq forced uniform via readfirstlane so all branches/loops around WMMA are
// scalar (no EXEC masking).
// ---------------------------------------------------------------------------
__global__ __launch_bounds__(128) void attention_k(
    const unsigned short* __restrict__ Q, const unsigned short* __restrict__ Kt,
    const unsigned short* __restrict__ Vt, unsigned short* __restrict__ O)
{
    __shared__ __align__(16) unsigned short Plds[4][16 * 32];

    const int lane  = threadIdx.x & 31;
    const int wave  = threadIdx.x >> 5;
    const int row16 = lane & 15;
    const int grp   = lane >> 4;
    const int tq = __builtin_amdgcn_readfirstlane(blockIdx.x * 4 + wave);  // wave-uniform SGPR
    const int h  = blockIdx.y, b = blockIdx.z;
    const int q0 = tq * 16;
    const long bh  = (long)b * SS * DDIM + h * DHH;          // Q/K/O base
    const long vbh = ((long)b * DDIM + h * DHH) * SS;        // Vt base

    unsigned short* Pw = &Plds[wave][0];

    Frag qa[2];
    {
        const unsigned short* qp = Q + bh + (long)(q0 + row16) * DDIM + 16 * grp;
#pragma unroll
        for (int c = 0; c < 2; ++c) {
            qa[c].q[0] = *(const v4u*)(qp + c * 32);
            qa[c].q[1] = *(const v4u*)(qp + c * 32 + 8);
        }
    }

    float mrow[8], lrow[8];
    v8f o[4] = {};
#pragma unroll
    for (int v = 0; v < 8; ++v) { mrow[v] = -1e30f; lrow[v] = 0.f; }
    const float scale = 0.125f; // 1/sqrt(64)

    for (int t2 = 0; t2 <= tq; t2 += 2) {
        float sa[8], sb[8];
        // ---- scores for key tile t2 ----
        {
            v8f sAv = {};
            const unsigned short* kp = Kt + bh + (long)(t2 * 16 + row16) * DDIM + 16 * grp;
#pragma unroll
            for (int c = 0; c < 2; ++c) {
                Frag kb;
                kb.q[0] = *(const v4u*)(kp + c * 32);
                kb.q[1] = *(const v4u*)(kp + c * 32 + 8);
                sAv = __builtin_amdgcn_wmma_f32_16x16x32_bf16(
                    false, qa[c].v, false, kb.v, (short)0, sAv, false, false);
            }
#pragma unroll
            for (int v = 0; v < 8; ++v) {
                float x = sAv[v] * scale;
                if (t2 == tq && row16 > v + 8 * grp) x = -1e9f;   // causal
                sa[v] = x;
            }
        }
        // ---- scores for key tile t2+1 (or fully masked pad) ----
        const int tB = t2 + 1;
        if (tB <= tq) {               // scalar branch (tq in SGPR)
            v8f sBv = {};
            const unsigned short* kp = Kt + bh + (long)(tB * 16 + row16) * DDIM + 16 * grp;
#pragma unroll
            for (int c = 0; c < 2; ++c) {
                Frag kb;
                kb.q[0] = *(const v4u*)(kp + c * 32);
                kb.q[1] = *(const v4u*)(kp + c * 32 + 8);
                sBv = __builtin_amdgcn_wmma_f32_16x16x32_bf16(
                    false, qa[c].v, false, kb.v, (short)0, sBv, false, false);
            }
#pragma unroll
            for (int v = 0; v < 8; ++v) {
                float x = sBv[v] * scale;
                if (tB == tq && row16 > v + 8 * grp) x = -1e9f;
                sb[v] = x;
            }
        } else {
#pragma unroll
            for (int v = 0; v < 8; ++v) sb[v] = -1e9f;
        }

        // ---- online softmax update ----
#pragma unroll
        for (int v = 0; v < 8; ++v) {
            float mx = fmaxf(sa[v], sb[v]);
            for (int d = 1; d < 16; d <<= 1) mx = fmaxf(mx, __shfl_xor(mx, d, 32));
            const float mnew  = fmaxf(mrow[v], mx);
            const float alpha = __expf(mrow[v] - mnew);
            mrow[v] = mnew;
            const float pa = __expf(sa[v] - mnew);
            const float pb = __expf(sb[v] - mnew);
            float rs = pa + pb;
            for (int d = 1; d < 16; d <<= 1) rs += __shfl_xor(rs, d, 32);
            lrow[v] = lrow[v] * alpha + rs;
#pragma unroll
            for (int j = 0; j < 4; ++j) o[j][v] *= alpha;
            Pw[(v + 8 * grp) * 32 + row16]      = f2bfbits(pa);
            Pw[(v + 8 * grp) * 32 + 16 + row16] = f2bfbits(pb);
        }
        asm volatile("" ::: "memory");   // keep per-wave DS program order

        // ---- P (C-layout) -> A-fragment via LDS (2 x ds_load_b128) ----
        Frag pf;
        pf.q[0] = *(const v4u*)&Pw[row16 * 32 + 16 * grp];
        pf.q[1] = *(const v4u*)&Pw[row16 * 32 + 16 * grp + 8];

        // ---- O += P * V over this 32-key chunk (Vt: b128 loads) ----
#pragma unroll
        for (int j = 0; j < 4; ++j) {
            Frag vb;
            const unsigned short* vp = Vt + vbh + (long)(j * 16 + row16) * SS + t2 * 16 + 16 * grp;
            vb.q[0] = *(const v4u*)(vp);
            vb.q[1] = *(const v4u*)(vp + 8);
            o[j] = __builtin_amdgcn_wmma_f32_16x16x32_bf16(
                false, pf.v, false, vb.v, (short)0, o[j], false, false);
        }
        asm volatile("" ::: "memory");
    }

    // ---- normalize and write bf16 attention output ----
#pragma unroll
    for (int v = 0; v < 8; ++v) {
        const float inv = 1.f / lrow[v];
        const long rowBase = bh + (long)(q0 + v + 8 * grp) * DDIM;
#pragma unroll
        for (int j = 0; j < 4; ++j)
            O[rowBase + j * 16 + row16] = f2bfbits(o[j][v] * inv);
    }
}

// ---------------------------------------------------------------------------
// Host launcher
// ---------------------------------------------------------------------------
extern "C" void kernel_launch(void* const* d_in, const int* in_sizes, int n_in,
                              void* d_out, int out_size, void* d_ws, size_t ws_size,
                              hipStream_t stream) {
    (void)in_sizes; (void)n_in; (void)out_size; (void)ws_size;
    const float* X   = (const float*)d_in[0];
    const float* g1  = (const float*)d_in[2];
    const float* o1  = (const float*)d_in[3];
    const float* g2  = (const float*)d_in[4];
    const float* o2  = (const float*)d_in[5];
    const float* Wq  = (const float*)d_in[6];
    const float* bq  = (const float*)d_in[7];
    const float* Wk  = (const float*)d_in[8];
    const float* bk  = (const float*)d_in[9];
    const float* Wv  = (const float*)d_in[10];
    const float* bv  = (const float*)d_in[11];
    const float* W0  = (const float*)d_in[12];
    const float* b0  = (const float*)d_in[13];
    const float* W1  = (const float*)d_in[14];
    const float* b1  = (const float*)d_in[15];
    const float* W2  = (const float*)d_in[16];
    const float* b2  = (const float*)d_in[17];
    float* Y = (float*)d_out;

    char* p = (char*)d_ws;
    auto alloc = [&](size_t bytes) {
        char* r = p; p += (bytes + 255) & ~(size_t)255; return r;
    };
    const long M = (long)BB * SS;                 // 4096 rows
    unsigned short* WqT  = (unsigned short*)alloc((size_t)DDIM * DDIM * 2);
    unsigned short* WkT  = (unsigned short*)alloc((size_t)DDIM * DDIM * 2);
    unsigned short* WvT  = (unsigned short*)alloc((size_t)DDIM * DDIM * 2);
    unsigned short* W0T  = (unsigned short*)alloc((size_t)DDIM * DDIM * 2);
    unsigned short* W1T  = (unsigned short*)alloc((size_t)DDIM * DFF * 2);
    unsigned short* W2T  = (unsigned short*)alloc((size_t)DFF * DDIM * 2);
    unsigned short* XnB  = (unsigned short*)alloc((size_t)M * DDIM * 2);
    unsigned short* Qb   = (unsigned short*)alloc((size_t)M * DDIM * 2);
    unsigned short* Kb   = (unsigned short*)alloc((size_t)M * DDIM * 2);
    unsigned short* VtB  = (unsigned short*)alloc((size_t)M * DDIM * 2);  // [B][H*DH][S]
    unsigned short* AttB = (unsigned short*)alloc((size_t)M * DDIM * 2);
    float*          X1   = (float*)alloc((size_t)M * DDIM * 4);
    unsigned short* HidB = (unsigned short*)alloc((size_t)M * DFF * 2);

    {
        const long nw = (long)DDIM * DDIM;
        cvt_headw_t<<<(unsigned)((nw + 255) / 256), 256, 0, stream>>>(Wq, WqT);
        cvt_headw_t<<<(unsigned)((nw + 255) / 256), 256, 0, stream>>>(Wk, WkT);
        cvt_headw_t<<<(unsigned)((nw + 255) / 256), 256, 0, stream>>>(Wv, WvT);
        cvt_w_t<<<(unsigned)((nw + 255) / 256), 256, 0, stream>>>(W0, W0T, DDIM, DDIM);
        const long nf = (long)DDIM * DFF;
        cvt_w_t<<<(unsigned)((nf + 255) / 256), 256, 0, stream>>>(W1, W1T, DDIM, DFF);
        cvt_w_t<<<(unsigned)((nf + 255) / 256), 256, 0, stream>>>(W2, W2T, DFF, DDIM);
    }

    layernorm_k<<<(unsigned)M, 256, 0, stream>>>(X, g1, o1, XnB);

    dim3 gD(DDIM / 64, (unsigned)(M / 128));
    gemm_tn<0,0,0,1,0><<<gD, 128, 0, stream>>>(XnB, WqT, bq, nullptr, nullptr, Qb, nullptr,
                                               (int)M, DDIM, DDIM);
    gemm_tn<0,0,0,1,0><<<gD, 128, 0, stream>>>(XnB, WkT, bk, nullptr, nullptr, Kb, nullptr,
                                               (int)M, DDIM, DDIM);
    gemm_tn<0,0,0,0,1><<<gD, 128, 0, stream>>>(XnB, WvT, bv, nullptr, nullptr, nullptr, VtB,
                                               (int)M, DDIM, DDIM);

    attention_k<<<dim3(SS / 64, HH, BB), 128, 0, stream>>>(Qb, Kb, VtB, AttB);

    gemm_tn<0,1,1,0,0><<<gD, 128, 0, stream>>>(AttB, W0T, b0, X, X1, nullptr, nullptr,
                                               (int)M, DDIM, DDIM);

    layernorm_k<<<(unsigned)M, 256, 0, stream>>>(X1, g2, o2, XnB);

    dim3 gF1(DFF / 64, (unsigned)(M / 128));
    gemm_tn<1,0,0,1,0><<<gF1, 128, 0, stream>>>(XnB, W1T, b1, nullptr, nullptr, HidB, nullptr,
                                                (int)M, DFF, DDIM);
    gemm_tn<0,1,1,0,0><<<gD, 128, 0, stream>>>(HidB, W2T, b2, X1, Y, nullptr, nullptr,
                                               (int)M, DDIM, DFF);
}
